// LSTM_42803644072363
// MI455X (gfx1250) — compile-verified, matrix operand order checked by reading, not run on hardware
//
#include <hip/hip_runtime.h>
#include <hip/hip_bf16.h>

// 2-layer LSTM, persistent-kernel formulation for MI455X (gfx1250).
// 16 workgroups x 256 threads (8 wave32 waves). Each WG owns a 16-unit slice
// of the hidden state (64 of the 1024 gate columns), keeps its bf16 weight
// slices resident in LDS (staged by the Tensor Data Mover with hardware
// padding for bank-conflict-free b128 reads), and the grid synchronizes twice
// per timestep via a global arrive counter while h0/h1 ping-pong through L2.

#define B_   32
#define T_   2048
#define H_   256
#define IN_  256
#define NWG  16
#define NTHR 256

// LDS layout (bytes)
#define LDS_W0    0                 // 64 rows * 520 bf16 = 66560 B
#define LDS_W1    66560             // 66560 B
#define LDS_GATES 133120            // 4*32*16 f32 = 8192 B
#define LDS_C0    141312            // 512 f32 = 2048 B
#define LDS_C1    143360            // 2048 B
#define LDS_BI0   145408            // 64 f32 = 256 B
#define LDS_BI1   145664            // 256 B
#define LDS_BYTES 145920
#define WROW      520               // padded bf16 row stride (260 dwords -> lane banks +4)

// workspace layout (bytes)
#define WS_W0T 0u                   // 1024*512 bf16 = 1 MiB (W0 transposed, n-major)
#define WS_W1T 1048576u             // 1 MiB
#define WS_H0G 2097152u             // 2 * 32*256 f32 ping-pong = 64 KiB
#define WS_H1G 2162688u             // 64 KiB
#define WS_CNT 2228224u             // barrier counter

typedef __bf16 bf16_t;
typedef bf16_t v16bf __attribute__((ext_vector_type(16)));
typedef float  v8f   __attribute__((ext_vector_type(8)));
typedef float  v4f   __attribute__((ext_vector_type(4)));
typedef unsigned int v4u __attribute__((ext_vector_type(4)));
typedef int    v4i   __attribute__((ext_vector_type(4)));
typedef int    v8i   __attribute__((ext_vector_type(8)));

union Frag { v16bf v; bf16_t e[16]; v4u q[2]; };

__device__ __forceinline__ void cvt8(const float* __restrict__ s, bf16_t* d) {
  v4f a = *(const v4f*)(s);
  v4f b = *(const v4f*)(s + 4);
#pragma unroll
  for (int i = 0; i < 4; ++i) { d[i] = (bf16_t)a[i]; d[4 + i] = (bf16_t)b[i]; }
}

__device__ __forceinline__ float sigm(float v) { return 1.0f / (1.0f + __expf(-v)); }

__device__ __forceinline__ void grid_barrier(unsigned* cnt, unsigned target) {
  __syncthreads();
  if (threadIdx.x == 0) {
    __hip_atomic_fetch_add(cnt, 1u, __ATOMIC_RELEASE, __HIP_MEMORY_SCOPE_AGENT);
    while (__hip_atomic_load(cnt, __ATOMIC_ACQUIRE, __HIP_MEMORY_SCOPE_AGENT) < target)
      __builtin_amdgcn_s_sleep(2);
  }
  __syncthreads();
}

// Full K=512 tile: two INDEPENDENT accumulator chains (arow0 covers K=0..255,
// arow1 covers K=256..511) interleaved so the scheduler can hide the
// WMMA->VALU A/B-reuse hazard slots with cvt/load work instead of v_nops.
__device__ __forceinline__ v8f wmma_k512(const float* __restrict__ arow0,
                                         const float* __restrict__ arow1,
                                         const bf16_t* __restrict__ wrow, int kup) {
  v8f a0 = {0.f, 0.f, 0.f, 0.f, 0.f, 0.f, 0.f, 0.f};
  v8f a1 = {0.f, 0.f, 0.f, 0.f, 0.f, 0.f, 0.f, 0.f};
#pragma unroll
  for (int kb = 0; kb < 8; ++kb) {
    const int k0 = kb * 32 + kup;
    Frag fa0, fa1, fb0, fb1;
    cvt8(arow0 + k0, fa0.e);
    cvt8(arow0 + k0 + 16, fa0.e + 8);
    cvt8(arow1 + k0, fa1.e);
    cvt8(arow1 + k0 + 16, fa1.e + 8);
    const bf16_t* wp0 = wrow + k0;
    const bf16_t* wp1 = wrow + 256 + k0;
    fb0.q[0] = *(const v4u*)(wp0); fb0.q[1] = *(const v4u*)(wp0 + 16);
    fb1.q[0] = *(const v4u*)(wp1); fb1.q[1] = *(const v4u*)(wp1 + 16);
    a0 = __builtin_amdgcn_wmma_f32_16x16x32_bf16(false, fa0.v, false, fb0.v,
                                                 (short)0, a0, false, false);
    a1 = __builtin_amdgcn_wmma_f32_16x16x32_bf16(false, fa1.v, false, fb1.v,
                                                 (short)0, a1, false, false);
  }
  return a0 + a1;
}

// One-shot: transpose + convert W (512 x 1024, k-major) -> Wt (1024 x 512 bf16, n-major)
__global__ void lstm_prep_weights(const float* __restrict__ W0,
                                  const float* __restrict__ W1,
                                  bf16_t* __restrict__ w0t,
                                  bf16_t* __restrict__ w1t) {
  int idx = blockIdx.x * blockDim.x + threadIdx.x;   // [0, 1024*512)
  int n = idx >> 9;
  int k = idx & 511;
  w0t[idx] = (bf16_t)W0[k * 1024 + n];
  w1t[idx] = (bf16_t)W1[k * 1024 + n];
}

// Stage one 16-row x 512-col bf16 chunk into LDS via the Tensor Data Mover.
// pad_enable inserts 4 DWORDs (8 bf16) after every 256 DWORDs (512 bf16), so
// rows land at the 520-element stride the GEMM expects — padding done by HW.
__device__ __forceinline__ void tdm_load_chunk(const bf16_t* gsrc, unsigned lds_off) {
#if __has_builtin(__builtin_amdgcn_tensor_load_to_lds)
  const unsigned long long ga = (unsigned long long)(uintptr_t)gsrc;
  v4u g0;
  g0[0] = 1u;                                          // count = 1, user mode
  g0[1] = lds_off;                                     // lds_addr (bytes)
  g0[2] = (unsigned)(ga & 0xFFFFFFFFull);              // global_addr[31:0]
  g0[3] = (unsigned)((ga >> 32) & 0x1FFFFFFull)        // global_addr[56:32]
        | (2u << 30);                                  // type = 2 ("image")
  v8i g1;
  g1[0] = (1 << 16)        // data_size = 2 bytes
        | (1 << 20)        // pad_enable
        | (7 << 22)        // pad_interval: 256 dwords
        | (3 << 25);       // pad_amount: 4 dwords
  g1[1] = (int)(512u << 16);   // tensor_dim0 = 512 (row length, elements)
  g1[2] = (int)(16u << 16);    // tensor_dim1 = 16 (rows)
  g1[3] = (int)(512u << 16);   // tile_dim0 = 512
  g1[4] = 16;                  // tile_dim1 = 16
  g1[5] = 512;                 // tensor_dim0_stride = 512
  g1[6] = 0;
  g1[7] = 0;
  v4i z4 = {0, 0, 0, 0};
#if defined(__clang_major__) && __clang_major__ >= 23
  v8i z8 = {0, 0, 0, 0, 0, 0, 0, 0};
  __builtin_amdgcn_tensor_load_to_lds(g0, g1, z4, z4, z8, 0);
#else
  __builtin_amdgcn_tensor_load_to_lds(g0, g1, z4, z4, 0);
#endif
#else
  (void)gsrc; (void)lds_off;
#endif
}

__global__ void __launch_bounds__(NTHR) lstm_persistent(
    const float* __restrict__ x,
    const float* __restrict__ c0i, const float* __restrict__ h0i,
    const float* __restrict__ c1i, const float* __restrict__ h1i,
    const float* __restrict__ b0,  const float* __restrict__ b1,
    const bf16_t* __restrict__ w0t, const bf16_t* __restrict__ w1t,
    float* __restrict__ h0g, float* __restrict__ h1g,
    unsigned* cnt, float* __restrict__ out) {
  extern __shared__ char smem[];
  bf16_t* w0l   = (bf16_t*)(smem + LDS_W0);
  bf16_t* w1l   = (bf16_t*)(smem + LDS_W1);
  float*  gates = (float*)(smem + LDS_GATES);   // [4][32][16]
  float*  c0l   = (float*)(smem + LDS_C0);      // [32][16]
  float*  c1l   = (float*)(smem + LDS_C1);
  float*  bi0   = (float*)(smem + LDS_BI0);     // [4][16]
  float*  bi1   = (float*)(smem + LDS_BI1);

  const int wg   = blockIdx.x;          // 0..15 : hidden slice [16*wg, 16*wg+16)
  const int tid  = threadIdx.x;
  const int lane = tid & 31;            // wave32
  const int wave = tid >> 5;            // 0..7
  const int q    = wave >> 1;           // gate 0..3 (i,j,f,o)
  const int m    = wave & 1;            // batch half
  const int col  = lane & 15;
  const int kup  = (lane >> 4) << 3;    // +8 K-offset (and C row-half) for lanes 16..31
  const int row  = m * 16 + col;        // batch row for the A fragment

  // ---- stage weight slices into LDS (reused 2048x) ----
#if __has_builtin(__builtin_amdgcn_tensor_load_to_lds)
  if (wave == 0) {                      // one wave issues the 8 TDM descriptors
#pragma unroll
    for (int qq = 0; qq < 4; ++qq) {
      const size_t gn0 = (size_t)(qq * 256 + wg * 16) * 512;
      tdm_load_chunk(w0t + gn0, LDS_W0 + (unsigned)(qq * 16) * (WROW * 2));
      tdm_load_chunk(w1t + gn0, LDS_W1 + (unsigned)(qq * 16) * (WROW * 2));
    }
    __builtin_amdgcn_s_wait_tensorcnt(0);
  }
#else
  for (int i = tid; i < 4096; i += NTHR) {      // 64 rows * 64 x b128 per matrix
    const int r  = i >> 6;
    const int cc = i & 63;
    const int gn = (r >> 4) * 256 + wg * 16 + (r & 15);
    *((v4u*)(w0l + r * WROW) + cc) = ((const v4u*)(w0t + (size_t)gn * 512))[cc];
    *((v4u*)(w1l + r * WROW) + cc) = ((const v4u*)(w1t + (size_t)gn * 512))[cc];
  }
#endif
  if (tid < 64) {
    const int gn = (tid >> 4) * 256 + wg * 16 + (tid & 15);
    bi0[tid] = b0[gn];
    bi1[tid] = b1[gn];
  }
  // initial states: cell slices to LDS, h slices to ping buffer 0
  for (int e = tid; e < 512; e += NTHR) {
    const int bb = e >> 4, u = e & 15, gcol = wg * 16 + u;
    c0l[e] = c0i[bb * 256 + gcol];
    c1l[e] = c1i[bb * 256 + gcol];
    h0g[bb * 256 + gcol] = h0i[bb * 256 + gcol];
    h1g[bb * 256 + gcol] = h1i[bb * 256 + gcol];
  }
  unsigned bar = 1;
  grid_barrier(cnt, bar * NWG);

  const bf16_t* w0row = w0l + (q * 16 + col) * WROW;   // this lane's B column
  const bf16_t* w1row = w1l + (q * 16 + col) * WROW;

  for (int t = 0; t < T_; ++t) {
    const float* h0r = h0g + (t & 1) * 8192;
    float*       h0w = h0g + ((t + 1) & 1) * 8192;
    const float* h1r = h1g + (t & 1) * 8192;
    float*       h1w = h1g + ((t + 1) & 1) * 8192;

    // ---- layer 0: gates = [x_t , h0] @ W0-slice ----
    {
      const float* xrow = x + ((size_t)row * T_ + t) * IN_;
      const float* hrow = h0r + row * 256;
      v8f acc = wmma_k512(xrow, hrow, w0row, kup);
      if (t + 1 < T_) __builtin_prefetch(xrow + IN_, 0, 1);  // next timestep's x row
#pragma unroll
      for (int r = 0; r < 8; ++r)
        gates[q * 512 + (m * 16 + kup + r) * 16 + col] = acc[r];
    }
    __syncthreads();
    for (int e = tid; e < 512; e += NTHR) {
      const int bb = e >> 4, u = e & 15;
      const float gi = gates[0 * 512 + e] + bi0[u];
      const float gj = gates[1 * 512 + e] + bi0[16 + u];
      const float gf = gates[2 * 512 + e] + bi0[32 + u];
      const float go = gates[3 * 512 + e] + bi0[48 + u];
      const float cn = c0l[e] * sigm(gf) + sigm(gi) * tanhf(gj);   // FORGET_BIAS = 0
      c0l[e] = cn;
      h0w[bb * 256 + wg * 16 + u] = tanhf(cn) * sigm(go);
    }
    ++bar; grid_barrier(cnt, bar * NWG);   // h0(t+1) visible device-wide

    // ---- layer 1: gates = [h0_new , h1_old] @ W1-slice ----
    {
      const float* h0row = h0w + row * 256;
      const float* h1row = h1r + row * 256;
      v8f acc = wmma_k512(h0row, h1row, w1row, kup);
#pragma unroll
      for (int r = 0; r < 8; ++r)
        gates[q * 512 + (m * 16 + kup + r) * 16 + col] = acc[r];
    }
    __syncthreads();
    for (int e = tid; e < 512; e += NTHR) {
      const int bb = e >> 4, u = e & 15;
      const float gi = gates[0 * 512 + e] + bi1[u];
      const float gj = gates[1 * 512 + e] + bi1[16 + u];
      const float gf = gates[2 * 512 + e] + bi1[32 + u];
      const float go = gates[3 * 512 + e] + bi1[48 + u];
      const float cn = c1l[e] * sigm(gf) + sigm(gi) * tanhf(gj);
      const float hn = tanhf(cn) * sigm(go);
      c1l[e] = cn;
      h1w[bb * 256 + wg * 16 + u] = hn;
      out[((size_t)bb * T_ + t) * H_ + wg * 16 + u] = hn;   // lstm_out[b][t][h]
    }
    ++bar; grid_barrier(cnt, bar * NWG);   // h1(t+1) visible device-wide
  }

  // ---- final states: out = [lstm_out | c0f | h0f | c1f | h1f] ----
  {
    const float* h0f = h0g + (T_ & 1) * 8192;
    const float* h1f = h1g + (T_ & 1) * 8192;
    const size_t base = (size_t)B_ * T_ * H_;
    for (int e = tid; e < 512; e += NTHR) {
      const int bb = e >> 4, u = e & 15, gcol = wg * 16 + u;
      out[base + 0 * 8192 + bb * 256 + gcol] = c0l[e];
      out[base + 1 * 8192 + bb * 256 + gcol] = h0f[bb * 256 + gcol];
      out[base + 2 * 8192 + bb * 256 + gcol] = c1l[e];
      out[base + 3 * 8192 + bb * 256 + gcol] = h1f[bb * 256 + gcol];
    }
  }
}

extern "C" void kernel_launch(void* const* d_in, const int* in_sizes, int n_in,
                              void* d_out, int out_size, void* d_ws, size_t ws_size,
                              hipStream_t stream) {
  (void)in_sizes; (void)n_in; (void)out_size; (void)ws_size;
  const float* x  = (const float*)d_in[0];
  const float* c0 = (const float*)d_in[1];
  const float* h0 = (const float*)d_in[2];
  const float* c1 = (const float*)d_in[3];
  const float* h1 = (const float*)d_in[4];
  const float* W0 = (const float*)d_in[5];
  const float* b0 = (const float*)d_in[6];
  const float* W1 = (const float*)d_in[7];
  const float* b1 = (const float*)d_in[8];
  float* out = (float*)d_out;

  char* ws = (char*)d_ws;
  bf16_t*   w0t = (bf16_t*)(ws + WS_W0T);
  bf16_t*   w1t = (bf16_t*)(ws + WS_W1T);
  float*    h0g = (float*)(ws + WS_H0G);
  float*    h1g = (float*)(ws + WS_H1G);
  unsigned* cnt = (unsigned*)(ws + WS_CNT);

  hipMemsetAsync(cnt, 0, sizeof(unsigned), stream);   // deterministic per-launch reset
  lstm_prep_weights<<<2048, 256, 0, stream>>>(W0, W1, w0t, w1t);

  static int lds_cfg = [] {
    hipFuncSetAttribute(reinterpret_cast<const void*>(lstm_persistent),
                        hipFuncAttributeMaxDynamicSharedMemorySize, LDS_BYTES);
    return 0;
  }();
  (void)lds_cfg;

  lstm_persistent<<<NWG, NTHR, LDS_BYTES, stream>>>(
      x, c0, h0, c1, h1, b0, b1, w0t, w1t, h0g, h1g, cnt, out);
}
